// VQEmbeddingEMA_46926812676247
// MI455X (gfx1250) — matrix operand order repaired and checked by reading, not compiled, not run on hardware
//
#include <hip/hip_runtime.h>
#include <hip/hip_bf16.h>

typedef __attribute__((ext_vector_type(16))) __bf16 v16bf;
typedef __attribute__((ext_vector_type(8)))  __bf16 v8bf;
typedef __attribute__((ext_vector_type(8)))  float  v8f;

#define NUM_CODE 1024
#define CODE_DIM 256
#define BATCH 16
#define TDIM 4096
#define NTOK (BATCH * TDIM)                    /* 65536 tokens */
#define QOFF ((size_t)BATCH * CODE_DIM * TDIM) /* 16777216 */
#define XPAD 264                               /* bf16 row stride, breaks LDS bank conflicts */
#define EPS_F 1.1920928955078125e-07f

static __device__ __forceinline__ __bf16 f2bf(float f) {
  union { float f; unsigned u; } v; v.f = f;
  unsigned r = v.u + 0x7FFFu + ((v.u >> 16) & 1u);   // round-to-nearest-even
  unsigned short h = (unsigned short)(r >> 16);
  union { unsigned short s; __bf16 b; } o; o.s = h;
  return o.b;
}

// ---------------------------------------------------------------------------
// Kernel 1: codebook -> bf16, per-code squared norm (fp32), zero counts.
// 1024 blocks x 256 threads (one block per code).
// ---------------------------------------------------------------------------
__global__ void vq_prep(const float* __restrict__ emb, __bf16* __restrict__ embB,
                        float* __restrict__ enorm, float* __restrict__ counts) {
  const int code = blockIdx.x;
  const int tid = threadIdx.x;
  const float v = emb[(size_t)code * CODE_DIM + tid];
  embB[(size_t)code * CODE_DIM + tid] = f2bf(v);
  float s = v * v;
#pragma unroll
  for (int off = 16; off > 0; off >>= 1) s += __shfl_xor(s, off, 32);
  __shared__ float red[8];
  if ((tid & 31) == 0) red[tid >> 5] = s;
  __syncthreads();
  if (tid == 0) {
    float t = 0.f;
#pragma unroll
    for (int i = 0; i < 8; ++i) t += red[i];
    enorm[code] = t;
    counts[code] = 0.f;
  }
}

// ---------------------------------------------------------------------------
// Kernel 2: bf16 WMMA distance GEMM + fused argmin over 1024 codes.
// 1024 blocks x 128 threads (4 waves); 64 tokens per block, 16 per wave.
// Double-buffered codebook tiles in LDS with register prefetch so the
// global-load latency of the next tile hides behind the WMMA chain.
// ---------------------------------------------------------------------------
__global__ void __launch_bounds__(128)
vq_argmin(const float* __restrict__ input, const __bf16* __restrict__ embB,
          const float* __restrict__ enorm, int* __restrict__ idxout) {
  __shared__ __bf16 Xs[64 * XPAD];      // 64 tokens x 256 dims (padded)
  __shared__ __bf16 Es[2][16 * XPAD];   // double-buffered 16-code tiles

  const int tid  = threadIdx.x;
  const int lane = tid & 31;
  const int wv   = tid >> 5;

  const int tokbase = blockIdx.x * 64;        // 64 tokens, never crosses a batch
  const int b  = tokbase >> 12;               // tokbase / 4096
  const int t0 = tokbase & 4095;
  const float* inb = input + (size_t)b * CODE_DIM * TDIM + t0;

  // Stage X tile: input[b, k, t0..t0+63] -> Xs[token][k] (bf16). Coalesced
  // float2 reads: one wave covers one k-row (64 floats) per iteration.
  for (int p = tid; p < (64 * CODE_DIM) / 2; p += 128) {
    const int k  = p >> 5;
    const int tt = (p & 31) * 2;
    const float2 v = *(const float2*)(inb + (size_t)k * TDIM + tt);
    Xs[tt * XPAD + k]       = f2bf(v.x);
    Xs[(tt + 1) * XPAD + k] = f2bf(v.y);
  }

  // Prologue: stage code tile 0 into buffer 0 (same barrier as X staging).
  {
    const int code = tid >> 5;            // tid in [0,128): 4 chunks/thread below
#pragma unroll
    for (int j = 0; j < 4; ++j) {
      const int c   = tid + j * 128;      // 512 chunks of 8 bf16
      const int cd  = c >> 5;
      const int off = (c & 31) * 8;
      *(uint4*)(&Es[0][cd * XPAD + off]) =
          *(const uint4*)(embB + (size_t)cd * CODE_DIM + off);
    }
    (void)code;
  }
  __syncthreads();

  // A-fragment layout (16-bit A 16x32, wave32): lane[3:0]=M row, lane[4]
  // selects K-halves {kh*8..kh*8+7, 16+kh*8..16+kh*8+7}.
  const int m     = lane & 15;   // A row / B column / D column (same lane id)
  const int khalf = lane >> 4;
  const int myrow = wv * 16 + m;

  v16bf A[8];
#pragma unroll
  for (int s = 0; s < 8; ++s) {
    const int kb = s * 32 + khalf * 8;
    const v8bf lo = *(const v8bf*)(&Xs[myrow * XPAD + kb]);
    const v8bf hi = *(const v8bf*)(&Xs[myrow * XPAD + kb + 16]);
    A[s] = __builtin_shufflevector(lo, hi, 0, 1, 2, 3, 4, 5, 6, 7,
                                           8, 9, 10, 11, 12, 13, 14, 15);
  }

  float best[8];
  int   bidx[8];
#pragma unroll
  for (int r = 0; r < 8; ++r) { best[r] = 3.4e38f; bidx[r] = 0; }

  for (int ct = 0; ct < 64; ++ct) {
    const int buf = ct & 1;

    // Prefetch next code tile (16 codes x 256 bf16) into registers: the
    // global-load latency overlaps the WMMA chain below.
    uint4 pf[4];
    if (ct < 63) {
#pragma unroll
      for (int j = 0; j < 4; ++j) {
        const int c   = tid + j * 128;
        const int cd  = c >> 5;
        const int off = (c & 31) * 8;
        pf[j] = *(const uint4*)(embB + ((size_t)((ct + 1) * 16 + cd)) * CODE_DIM + off);
      }
    }

    // Load all 8 B-fragments first so the 16 ds_load_b128 pipeline instead
    // of stalling dscnt==0 before each WMMA.
    // B layout (32x16 bf16): lane[3:0]=N column, lane[4] selects K-halves
    // {kh*8..kh*8+7, 16+kh*8..16+kh*8+7}, matching the A fragment.
    v16bf Bf[8];
#pragma unroll
    for (int s = 0; s < 8; ++s) {
      const __bf16* ep = &Es[buf][m * XPAD + s * 32 + khalf * 8];
      const v8bf lo = *(const v8bf*)(ep);
      const v8bf hi = *(const v8bf*)(ep + 16);
      Bf[s] = __builtin_shufflevector(lo, hi, 0, 1, 2, 3, 4, 5, 6, 7,
                                              8, 9, 10, 11, 12, 13, 14, 15);
    }

    v8f acc = {0.f, 0.f, 0.f, 0.f, 0.f, 0.f, 0.f, 0.f};
#pragma unroll
    for (int s = 0; s < 8; ++s) {
      acc = __builtin_amdgcn_wmma_f32_16x16x32_bf16(
          /*neg_a=*/false, A[s], /*neg_b=*/false, Bf[s],
          /*c_mod=*/(short)0, acc, /*reuse_a=*/false, /*reuse_b=*/false);
    }

    // D layout: VGPR r holds rows M = r + khalf*8, column N = m.
    const float en = enorm[ct * 16 + m];
#pragma unroll
    for (int r = 0; r < 8; ++r) {
      const float sc = en - 2.0f * acc[r];      // ||x||^2 constant per row: dropped
      if (sc < best[r]) { best[r] = sc; bidx[r] = ct * 16 + m; }
    }

    // Commit prefetched tile into the other buffer, then one barrier/iter.
    if (ct < 63) {
#pragma unroll
      for (int j = 0; j < 4; ++j) {
        const int c   = tid + j * 128;
        const int cd  = c >> 5;
        const int off = (c & 31) * 8;
        *(uint4*)(&Es[buf ^ 1][cd * XPAD + off]) = pf[j];
      }
    }
    __syncthreads();
  }

  // Reduce (min, argmin) across the 16 lanes sharing each token row.
#pragma unroll
  for (int off = 1; off < 16; off <<= 1) {
#pragma unroll
    for (int r = 0; r < 8; ++r) {
      const float ov = __shfl_xor(best[r], off, 32);
      const int   oi = __shfl_xor(bidx[r], off, 32);
      if (ov < best[r] || (ov == best[r] && oi < bidx[r])) {
        best[r] = ov; bidx[r] = oi;
      }
    }
  }
  if (m == 0) {
    const int base = tokbase + wv * 16 + khalf * 8;
#pragma unroll
    for (int r = 0; r < 8; ++r) idxout[base + r] = bidx[r];
  }
}

// ---------------------------------------------------------------------------
// Kernel 3: counts (float atomics) + indices output.
// ---------------------------------------------------------------------------
__global__ void vq_scatter(const int* __restrict__ idxws, float* __restrict__ counts,
                           float* __restrict__ out_idx) {
  const int tok = blockIdx.x * blockDim.x + threadIdx.x;
  const int idx = idxws[tok];
  atomicAdd(&counts[idx], 1.0f);
  out_idx[tok] = (float)idx;
}

// ---------------------------------------------------------------------------
// Kernel 4: quantized = embedding[indices], written back as [B, N, T].
// ---------------------------------------------------------------------------
__global__ void vq_gather(const int* __restrict__ idxws, const float* __restrict__ emb,
                          float* __restrict__ outq) {
  const size_t lin = (size_t)blockIdx.x * blockDim.x + threadIdx.x;
  const int t = (int)(lin & 4095);
  const int n = (int)((lin >> 12) & 255);
  const int b = (int)(lin >> 20);
  const int idx = idxws[b * TDIM + t];
  outq[lin] = emb[(size_t)idx * CODE_DIM + n];
}

// ---------------------------------------------------------------------------
// Kernel 5: perplexity = exp(-sum p log(p + eps)), single block.
// ---------------------------------------------------------------------------
__global__ void vq_perplexity(const float* __restrict__ counts, float* __restrict__ outp) {
  const int tid = threadIdx.x;
  float s = 0.f;
  for (int i = tid; i < NUM_CODE; i += 256) {
    const float p = counts[i] * (1.0f / (float)NTOK);
    s -= p * logf(p + EPS_F);
  }
#pragma unroll
  for (int off = 16; off > 0; off >>= 1) s += __shfl_xor(s, off, 32);
  __shared__ float red[8];
  if ((tid & 31) == 0) red[tid >> 5] = s;
  __syncthreads();
  if (tid == 0) {
    float t = 0.f;
#pragma unroll
    for (int i = 0; i < 8; ++i) t += red[i];
    *outp = expf(t);
  }
}

extern "C" void kernel_launch(void* const* d_in, const int* in_sizes, int n_in,
                              void* d_out, int out_size, void* d_ws, size_t ws_size,
                              hipStream_t stream) {
  (void)in_sizes; (void)n_in; (void)out_size; (void)ws_size;
  const float* input = (const float*)d_in[0];   // [16, 256, 4096] f32
  const float* emb   = (const float*)d_in[1];   // [1024, 256] f32
  float* out = (float*)d_out;                   // quantized | indices | perplexity

  // Workspace carve-out (~776 KB total).
  char* ws = (char*)d_ws;
  __bf16* embB  = (__bf16*)ws;                         // 1024*256*2 = 524288 B
  float*  enorm = (float*)(ws + 524288);               // 4096 B
  float*  counts = (float*)(ws + 524288 + 4096);       // 4096 B
  int*    idxws  = (int*)(ws + 524288 + 8192);         // 65536*4 = 262144 B

  vq_prep<<<NUM_CODE, CODE_DIM, 0, stream>>>(emb, embB, enorm, counts);
  vq_argmin<<<NTOK / 64, 128, 0, stream>>>(input, embB, enorm, idxws);
  vq_scatter<<<NTOK / 256, 256, 0, stream>>>(idxws, counts, out + QOFF);
  vq_gather<<<(unsigned)(QOFF / 256), 256, 0, stream>>>(idxws, emb, out);
  vq_perplexity<<<1, 256, 0, stream>>>(counts, out + QOFF + NTOK);
}
